// FinalMultiHeadAttention_90787018703088
// MI455X (gfx1250) — matrix operand order repaired and checked by reading, not compiled, not run on hardware
//
#include <hip/hip_runtime.h>
#include <math.h>

// ---------------------------------------------------------------------------
// MI455X (gfx1250) multi-head attention.  B=8, N=1024, C=1024, H=16, D=64.
//
//  Phase 0: split-convert x -> bf16 hi/lo planes; transpose-split w_qkv/w_proj
//  Phase 1: qkv = x @ w_qkv + b_qkv       bf16x3 WMMA, LDS-staged A slab
//  Phase 2: flash attention               exact fp32 WMMA (v_wmma_f32_16x16x4_f32)
//  Phase 3: out = ctx @ w_proj + b_proj   bf16x3 WMMA, LDS-staged A slab
//
// bf16x3: A = Ah + Al (bf16 planes), A.B ~= Ah.Bh + Al.Bh + Ah.Bl, fp32 accum.
// GEMM blocks own a 64x128 tile: the shared 64-row A slab is double-buffered
// through LDS (8x fewer redundant global A loads than the per-wave scheme).
// ---------------------------------------------------------------------------

typedef float  v2f  __attribute__((ext_vector_type(2)));
typedef float  v4f  __attribute__((ext_vector_type(4)));
typedef float  v8f  __attribute__((ext_vector_type(8)));
typedef __bf16 v4bf __attribute__((ext_vector_type(4)));
typedef __bf16 v8bf __attribute__((ext_vector_type(8)));
typedef __bf16 v16bf __attribute__((ext_vector_type(16)));

__device__ __forceinline__ v8f wmma_f32(v8f c, v2f a, v2f b) {
  return __builtin_amdgcn_wmma_f32_16x16x4_f32(false, a, false, b, (short)0, c,
                                               false, false);
}
__device__ __forceinline__ v8f wmma_bf16(v8f c, v16bf a, v16bf b) {
  return __builtin_amdgcn_wmma_f32_16x16x32_bf16(false, a, false, b, (short)0,
                                                 c, false, false);
}
__device__ __forceinline__ v16bf cat8(v8bf a, v8bf b) {
  return __builtin_shufflevector(a, b, 0, 1, 2, 3, 4, 5, 6, 7, 8, 9, 10, 11,
                                 12, 13, 14, 15);
}

// ---------------------------------------------------------------------------
// Split-convert fp32 -> bf16 hi/lo planes (same layout). 4 elements/thread.
// ---------------------------------------------------------------------------
__global__ __launch_bounds__(256) void convert_split(
    const float* __restrict__ in, __bf16* __restrict__ hi,
    __bf16* __restrict__ lo, int count4) {
  const int i = blockIdx.x * 256 + threadIdx.x;
  if (i >= count4) return;
  v4f x = ((const v4f*)in)[i];
  v4bf h, l;
#pragma unroll
  for (int j = 0; j < 4; ++j) {
    const __bf16 hj = (__bf16)x[j];
    h[j] = hj;
    l[j] = (__bf16)(x[j] - (float)hj);
  }
  ((v4bf*)hi)[i] = h;
  ((v4bf*)lo)[i] = l;
}

// ---------------------------------------------------------------------------
// Transpose + split: W[K,N] fp32 -> Thi/Tlo[N,K] bf16. LDS 32x32 tiles.
// Grid: (N/32, K/32), block (32,8).
// ---------------------------------------------------------------------------
__global__ __launch_bounds__(256) void transpose_split(
    const float* __restrict__ W, __bf16* __restrict__ Thi,
    __bf16* __restrict__ Tlo, int K, int N) {
  __shared__ float tile[32][33];
  const int n0 = blockIdx.x * 32;
  const int k0 = blockIdx.y * 32;
#pragma unroll
  for (int j = 0; j < 32; j += 8) {
    const int k = k0 + threadIdx.y + j;
    tile[threadIdx.y + j][threadIdx.x] = W[(size_t)k * N + n0 + threadIdx.x];
  }
  __syncthreads();
#pragma unroll
  for (int j = 0; j < 32; j += 8) {
    const int n = n0 + threadIdx.y + j;
    const float x = tile[threadIdx.x][threadIdx.y + j];  // = W[k0+tx][n]
    const __bf16 h = (__bf16)x;
    Thi[(size_t)n * K + k0 + threadIdx.x] = h;
    Tlo[(size_t)n * K + k0 + threadIdx.x] = (__bf16)(x - (float)h);
  }
}

// ---------------------------------------------------------------------------
// bf16x3 GEMM: C[M,N] = (Ah+Al)[M,K] @ (Bh+Bl)[K,N] + bias, fp32 accumulate.
// B given transposed: Bt*[N,K].
// Block (8 waves) owns a 64x128 tile: wave wv -> columns [n0+16*wv, +16).
// Shared 64-row A slab (32-K step, hi+lo) double-buffered in LDS, padded to
// 40 bf16/row so 16B fragment reads cover all 64 banks.
// A-frag (16-bit A 16x32 layout): lane l (half hf) = row l%16,
//   runs K = [k+16*hf, +16) (contiguous, same as B)  -> 2x ds b128 per plane.
// B-frag: lane l = column n, K = [k+16*hf, +16) -> contiguous in Bt row.
// Requires M%64==0, N%128==0, K%32==0, blockDim==256.
// ---------------------------------------------------------------------------
#define SLAB_LD 40  // bf16 elements per LDS row (32 data + 8 pad)
__global__ __launch_bounds__(256) void gemm_bf16x3(
    const __bf16* __restrict__ Ahi, const __bf16* __restrict__ Alo,
    const __bf16* __restrict__ Bthi, const __bf16* __restrict__ Btlo,
    const float* __restrict__ bias, float* __restrict__ C, int M, int N,
    int K) {
  __shared__ __align__(16) __bf16 sA[2][2][64 * SLAB_LD];  // [buf][plane]

  const int lane = threadIdx.x & 31;
  const int wv = threadIdx.x >> 5;
  const int nblks = N >> 7;  // 128-wide column blocks
  const int mblk = blockIdx.x / nblks;
  const int nblk = blockIdx.x - mblk * nblks;
  const int m0 = mblk * 64;
  const int n0 = nblk * 128 + wv * 16;
  const int hf = lane >> 4;
  const int l16 = lane & 15;

  // cooperative A-slab copy: thread tid copies one 16B chunk of each plane
  const int crow = threadIdx.x >> 2;        // 0..63
  const int ckc = (threadIdx.x & 3) * 8;    // 0,8,16,24
  const size_t cgoff = (size_t)(m0 + crow) * K + ckc;
  const int cloff = crow * SLAB_LD + ckc;

  v8f acc[4] = {v8f{}, v8f{}, v8f{}, v8f{}};

  // prologue: slab for k=0 into buffer 0
  *(v8bf*)&sA[0][0][cloff] = *(const v8bf*)&Ahi[cgoff];
  *(v8bf*)&sA[0][1][cloff] = *(const v8bf*)&Alo[cgoff];
  __syncthreads();

  for (int k = 0, it = 0; k < K; k += 32, ++it) {
    const int cur = it & 1;
    // copy next slab into the other buffer (overlaps with compute below)
    if (k + 32 < K) {
      *(v8bf*)&sA[cur ^ 1][0][cloff] = *(const v8bf*)&Ahi[cgoff + k + 32];
      *(v8bf*)&sA[cur ^ 1][1][cloff] = *(const v8bf*)&Alo[cgoff + k + 32];
    }

    // B fragments (global, per-wave; prefetch next slab's lines)
    const size_t boff = (size_t)(n0 + l16) * K + k + 16 * hf;
    const v16bf bh = cat8(*(const v8bf*)&Bthi[boff],
                          *(const v8bf*)&Bthi[boff + 8]);
    const v16bf bl = cat8(*(const v8bf*)&Btlo[boff],
                          *(const v8bf*)&Btlo[boff + 8]);
    if (k + 32 < K) {
      __builtin_prefetch(&Bthi[boff + 32], 0, 1);
      __builtin_prefetch(&Btlo[boff + 32], 0, 1);
    }

    // A fragments from the LDS slab
    v16bf ah[4], al[4];
#pragma unroll
    for (int t = 0; t < 4; ++t) {
      const int ao = (t * 16 + l16) * SLAB_LD + 16 * hf;
      ah[t] = cat8(*(const v8bf*)&sA[cur][0][ao],
                   *(const v8bf*)&sA[cur][0][ao + 8]);
      al[t] = cat8(*(const v8bf*)&sA[cur][1][ao],
                   *(const v8bf*)&sA[cur][1][ao + 8]);
    }

    // 12 WMMAs, interleaved across subtiles to break accumulator chains
#pragma unroll
    for (int t = 0; t < 4; ++t) acc[t] = wmma_bf16(acc[t], ah[t], bh);
#pragma unroll
    for (int t = 0; t < 4; ++t) acc[t] = wmma_bf16(acc[t], al[t], bh);
#pragma unroll
    for (int t = 0; t < 4; ++t) acc[t] = wmma_bf16(acc[t], ah[t], bl);

    __syncthreads();  // slab copy + all reads of buf[cur] complete
  }

  const float bv = bias[n0 + l16];
#pragma unroll
  for (int t = 0; t < 4; ++t) {
#pragma unroll
    for (int r = 0; r < 8; ++r) {
      const int row = m0 + t * 16 + r + 8 * hf;  // f32 C-layout row mapping
      C[(size_t)row * N + n0 + l16] = acc[t][r] + bv;
    }
  }
}

// ---------------------------------------------------------------------------
// Flash attention, exact fp32 WMMA. qkv: [B*N, 3C], ctx: [B*N, C] (head-merged)
// One wave per (b, h, 16-row Q block); streams 64 key blocks of 16.
// ---------------------------------------------------------------------------
__global__ __launch_bounds__(256) void attn_flash_wmma(
    const float* __restrict__ qkv, float* __restrict__ ctx) {
  constexpr int Cdim = 1024, H = 16, D = 64, N = 1024;
  constexpr int ld = 3 * Cdim;

  __shared__ __align__(16) float sP[8][16 * 16];  // one 16x16 P tile per wave

  const int lane = threadIdx.x & 31;
  const int wv = threadIdx.x >> 5;
  const int wave = blockIdx.x * 8 + wv;
  const int qblk = wave & 63;
  const int h = (wave >> 6) & (H - 1);
  const int b = wave >> 10;
  const int hf = lane >> 4;
  const int l16 = lane & 15;

  const float* qbase = qkv + (size_t)(b * N) * ld + h * D;
  const float* kbase = qbase + Cdim;
  const float* vbase = qbase + 2 * Cdim;

  const float scale = 0.125f;  // 1/sqrt(64)
  v2f qf[16];
  {
    const float* qrow = qbase + (size_t)(qblk * 16 + l16) * ld;
#pragma unroll
    for (int s = 0; s < 16; ++s) {
      v2f t = *(const v2f*)&qrow[s * 4 + 2 * hf];
      qf[s].x = t.x * scale;
      qf[s].y = t.y * scale;
    }
  }

  v8f o[4] = {v8f{}, v8f{}, v8f{}, v8f{}};
  float mRun[8], lRun[8];
#pragma unroll
  for (int r = 0; r < 8; ++r) {
    mRun[r] = -__builtin_inff();
    lRun[r] = 0.0f;
  }

  float* myP = &sP[wv][0];

  for (int kb = 0; kb < N / 16; ++kb) {
    // S = (Q*scale) @ K^T
    v8f s = v8f{};
    const float* krow = kbase + (size_t)(kb * 16 + l16) * ld;
#pragma unroll
    for (int ks = 0; ks < 16; ++ks) {
      v2f bf = *(const v2f*)&krow[ks * 4 + 2 * hf];
      s = wmma_f32(s, qf[ks], bf);
    }

    // online softmax per row (each row lives on one 16-lane half)
    float p[8];
#pragma unroll
    for (int r = 0; r < 8; ++r) {
      float v = s[r];
      float mx = v;
#pragma unroll
      for (int m = 1; m < 16; m <<= 1) mx = fmaxf(mx, __shfl_xor(mx, m, 32));
      const float mNew = fmaxf(mRun[r], mx);
      const float corr = __expf(mRun[r] - mNew);
      const float pe = __expf(v - mNew);
      float rs = pe;
#pragma unroll
      for (int m = 1; m < 16; m <<= 1) rs += __shfl_xor(rs, m, 32);
      lRun[r] = lRun[r] * corr + rs;
      mRun[r] = mNew;
      p[r] = pe;
#pragma unroll
      for (int t = 0; t < 4; ++t) o[t][r] *= corr;
    }

    // stage P through LDS: C-layout -> A-layout (per-wave, DS is in-order)
#pragma unroll
    for (int r = 0; r < 8; ++r) myP[(r + 8 * hf) * 16 + l16] = p[r];
    __builtin_amdgcn_wave_barrier();

    // O += P @ V
    const float* vblk = vbase + (size_t)(kb * 16) * ld;
#pragma unroll
    for (int ks = 0; ks < 4; ++ks) {
      const int kk = ks * 4 + 2 * hf;
      v2f pf = *(const v2f*)&myP[l16 * 16 + kk];
#pragma unroll
      for (int t = 0; t < 4; ++t) {
        v2f bf;
        bf.x = vblk[(size_t)kk * ld + t * 16 + l16];
        bf.y = vblk[(size_t)(kk + 1) * ld + t * 16 + l16];
        o[t] = wmma_f32(o[t], pf, bf);
      }
    }
    __builtin_amdgcn_wave_barrier();
  }

  const int row0 = qblk * 16;
#pragma unroll
  for (int t = 0; t < 4; ++t) {
#pragma unroll
    for (int r = 0; r < 8; ++r) {
      const int row = row0 + r + 8 * hf;
      const float inv = 1.0f / lRun[r];
      ctx[(size_t)(b * N + row) * Cdim + h * D + t * 16 + l16] = o[t][r] * inv;
    }
  }
}

// ---------------------------------------------------------------------------
extern "C" void kernel_launch(void* const* d_in, const int* in_sizes, int n_in,
                              void* d_out, int out_size, void* d_ws,
                              size_t ws_size, hipStream_t stream) {
  (void)in_sizes; (void)n_in; (void)out_size; (void)ws_size;
  const float* x      = (const float*)d_in[0];  // [8192,1024]
  const float* w_qkv  = (const float*)d_in[1];  // [1024,3072]
  const float* b_qkv  = (const float*)d_in[2];  // [3072]
  const float* w_proj = (const float*)d_in[3];  // [1024,1024]
  const float* b_proj = (const float*)d_in[4];  // [1024]
  float* out = (float*)d_out;                   // [8192,1024]

  // workspace carve (all sizes 64B-multiples): ~185 MB total
  char* ws = (char*)d_ws;
  float* qkv = (float*)ws;          ws += (size_t)8192 * 3072 * 4;  // 100.7 MB
  float* ctx = (float*)ws;          ws += (size_t)8192 * 1024 * 4;  //  33.6 MB
  __bf16* pAhi = (__bf16*)ws;       ws += (size_t)8192 * 1024 * 2;  //  16.8 MB
  __bf16* pAlo = (__bf16*)ws;       ws += (size_t)8192 * 1024 * 2;  //  16.8 MB
  __bf16* wqkvThi = (__bf16*)ws;    ws += (size_t)3072 * 1024 * 2;  //   6.3 MB
  __bf16* wqkvTlo = (__bf16*)ws;    ws += (size_t)3072 * 1024 * 2;
  __bf16* wprojThi = (__bf16*)ws;   ws += (size_t)1024 * 1024 * 2;  //   2.1 MB
  __bf16* wprojTlo = (__bf16*)ws;   ws += (size_t)1024 * 1024 * 2;

  // Phase 0: one-time precision-split / transpose of operands
  transpose_split<<<dim3(3072 / 32, 1024 / 32), dim3(32, 8), 0, stream>>>(
      w_qkv, wqkvThi, wqkvTlo, 1024, 3072);
  transpose_split<<<dim3(1024 / 32, 1024 / 32), dim3(32, 8), 0, stream>>>(
      w_proj, wprojThi, wprojTlo, 1024, 1024);
  const int cnt4 = 8192 * 1024 / 4;  // 2,097,152
  convert_split<<<cnt4 / 256, 256, 0, stream>>>(x, pAhi, pAlo, cnt4);

  // Phase 1: qkv = x @ w_qkv + b_qkv   (bf16x3; 128 x 24 blocks of 64x128)
  gemm_bf16x3<<<128 * (3072 / 128), 256, 0, stream>>>(
      pAhi, pAlo, wqkvThi, wqkvTlo, b_qkv, qkv, 8192, 3072, 1024);

  // Phase 2: flash attention (8192 waves, exact fp32 WMMA)
  attn_flash_wmma<<<8192 / 8, 256, 0, stream>>>(qkv, ctx);

  // Phase 3: out = ctx @ w_proj + b_proj   (bf16x3, reuse A-plane buffers)
  convert_split<<<cnt4 / 256, 256, 0, stream>>>(ctx, pAhi, pAlo, cnt4);
  gemm_bf16x3<<<128 * (1024 / 128), 256, 0, stream>>>(
      pAhi, pAlo, wprojThi, wprojTlo, b_proj, out, 8192, 1024, 1024);
}
